// HGMM_46428596469829
// MI455X (gfx1250) — matrix operand order repaired and checked by reading, not compiled, not run on hardware
//
#include <hip/hip_runtime.h>
#include <hip/hip_bf16.h>
#include <math.h>

typedef __attribute__((ext_vector_type(16))) _Float16 v16h;
typedef __attribute__((ext_vector_type(8)))  _Float16 v8h;
typedef __attribute__((ext_vector_type(8)))  float    v8f;

#define HW    120
#define RTOT  14400            // = 120*120 = 25*576 : rows of every big GEMM
#define IMG_N (64 * 14400)     // elements of one (64,120,120) image
#define NCH   60               // chunks for the window-scan parallelization
#define LCH   240              // 60 * 240 = 14400

__device__ __forceinline__ float sigmoidf_(float x) { return 1.f / (1.f + __expf(-x)); }
__device__ __forceinline__ float softplusf_(float x) { return (x > 20.f) ? x : log1pf(__expf(x)); }

// ---------------------------------------------------------------------------
// WMMA GEMM: C[M,N](f32,ldc) = A[M,K](f16) * B[N,K]^T(f16), lda==ldb==K.
// One wave32 per 16x(16*NT) tile; A fragment reused by NT WMMAs.
// K compile-time -> fully unrolled, software-pipelined: fragments for k-step
// s+1 are issued before the WMMAs of step s, so s_wait_loadcnt covers loads
// launched a whole WMMA-group earlier (keeps the XDL pipe fed).
// revstore: store row' = M-1-row (yM reversed write into nf).
// Fragments: A/B use contiguous k = 16*half + s, C per CDNA5 ISA 7.12.2 (16-bit A 16x32, B 32x16, f32 C 16x16).
// ---------------------------------------------------------------------------
template <int NT, int K>
__global__ void gemm_wmma(const _Float16* __restrict__ A, const _Float16* __restrict__ B,
                          float* __restrict__ C, int M, int ldc, int revstore)
{
    constexpr int KS = K / 32;
    const int mt = blockIdx.x, nt0 = blockIdx.y * NT;
    const int t = threadIdx.x;           // 0..31
    const int half = t >> 4;             // lane half: selects K sub-blocks
    const int l16 = t & 15;
    const _Float16* Ap = A + (size_t)(mt * 16 + l16) * K;     // A row m
    const _Float16* Bp[NT];
#pragma unroll
    for (int j = 0; j < NT; ++j)
        Bp[j] = B + (size_t)((nt0 + j) * 16 + l16) * K;       // B row n
    __builtin_prefetch(Ap, 0, 0);                             // global_prefetch_b8

    // A frag: lanes 0-15 K=k0..k0+15, lanes 16-31 K=k0+16..k0+31
    auto loadA = [&](int k0) -> v16h {
        return *(const v16h*)(Ap + k0 + half * 16);
    };
    // B frag: lanes 0-15 K=k0..k0+15, lanes 16-31 K=k0+16..k0+31
    auto loadB = [&](int j, int k0) -> v16h {
        return *(const v16h*)(Bp[j] + k0 + half * 16);
    };

    v8f acc[NT];
#pragma unroll
    for (int j = 0; j < NT; ++j) acc[j] = (v8f){};

    v16h a_cur = loadA(0);
    v16h b_cur[NT];
#pragma unroll
    for (int j = 0; j < NT; ++j) b_cur[j] = loadB(j, 0);

#pragma unroll
    for (int s = 0; s < KS; ++s) {
        v16h a_nxt;
        v16h b_nxt[NT];
        if (s + 1 < KS) {                       // issue next fragments first
            a_nxt = loadA((s + 1) * 32);
#pragma unroll
            for (int j = 0; j < NT; ++j) b_nxt[j] = loadB(j, (s + 1) * 32);
        }
#pragma unroll
        for (int j = 0; j < NT; ++j)
            acc[j] = __builtin_amdgcn_wmma_f32_16x16x32_f16(false, a_cur, false, b_cur[j],
                                                            (short)0, acc[j], false, false);
        if (s + 1 < KS) {
            a_cur = a_nxt;
#pragma unroll
            for (int j = 0; j < NT; ++j) b_cur[j] = b_nxt[j];
        }
    }

#pragma unroll
    for (int j = 0; j < NT; ++j)
#pragma unroll
        for (int r = 0; r < 8; ++r) {   // D: VGPR r => row half*8+r, col = lane&15
            int row = mt * 16 + half * 8 + r;
            int srow = revstore ? (M - 1 - row) : row;
            C[(size_t)srow * ldc + (nt0 + j) * 16 + l16] = acc[j][r];
        }
}

// ---------------------------------------------------------------------------
// Small utility kernels
// ---------------------------------------------------------------------------
__global__ void k_f32_to_f16(const float* __restrict__ s, _Float16* __restrict__ d, int n)
{
    int i = blockIdx.x * blockDim.x + threadIdx.x;
    if (i < n) d[i] = (_Float16)s[i];
}

// xproj weights (8 groups of 36x128) -> f16 padded to 48x128 (zero rows 36..47)
__global__ void k_xproj_pad(const float* __restrict__ s, _Float16* __restrict__ d)
{
    int i = blockIdx.x * blockDim.x + threadIdx.x;
    if (i >= 8 * 48 * 128) return;
    int c = i & 127, row = (i >> 7) % 48, g = i / (48 * 128);
    d[i] = (row < 36) ? (_Float16)s[(g * 36 + row) * 128 + c] : (_Float16)0.f;
}

__global__ void k_transpose_img(const float* __restrict__ s, float* __restrict__ d)
{
    int i = blockIdx.x * blockDim.x + threadIdx.x;
    if (i >= IMG_N) return;
    int w = i % HW, h = (i / HW) % HW, c = i / (HW * HW);
    d[i] = s[(c * HW + w) * HW + h];
}

// LayerNorm over C=64 for the window path: row l, channel stride RTOT in image
__global__ void k_ln_window(const float* __restrict__ x, const float* __restrict__ g,
                            const float* __restrict__ b, _Float16* __restrict__ xn)
{
    int l = blockIdx.x * blockDim.x + threadIdx.x;
    if (l >= RTOT) return;
    float s = 0.f, ss = 0.f;
    for (int c = 0; c < 64; ++c) { float v = x[c * RTOT + l]; s += v; ss += v * v; }
    float mu = s * (1.f / 64.f);
    float inv = rsqrtf(ss * (1.f / 64.f) - mu * mu + 1e-5f);
    for (int c = 0; c < 64; ++c) {
        float v = (x[c * RTOT + l] - mu) * inv * g[c] + b[c];
        xn[(size_t)l * 64 + c] = (_Float16)v;
    }
}

// LayerNorm + 5x5 pixel-shuffle gather: row rho = (pi*5+pj)*576 + nh*24+nw
__global__ void k_ln_pixel(const float* __restrict__ x, const float* __restrict__ g,
                           const float* __restrict__ b, _Float16* __restrict__ xn)
{
    int r = blockIdx.x * blockDim.x + threadIdx.x;
    if (r >= RTOT) return;
    int pp = r / 576, l = r % 576;
    int nh = l / 24, nw = l % 24;
    int pi = pp / 5, pj = pp % 5;
    int base = (nh * 5 + pi) * HW + (nw * 5 + pj);
    float s = 0.f, ss = 0.f;
    for (int c = 0; c < 64; ++c) { float v = x[c * RTOT + base]; s += v; ss += v * v; }
    float mu = s * (1.f / 64.f);
    float inv = rsqrtf(ss * (1.f / 64.f) - mu * mu + 1e-5f);
    for (int c = 0; c < 64; ++c) {
        float v = (x[c * RTOT + base] - mu) * inv * g[c] + b[c];
        xn[(size_t)r * 64 + c] = (_Float16)v;
    }
}

// Depthwise causal (dir=0) / anti-causal flipped (dir=1) conv k=4 + SiLU.
// dir=1 implements the reversed-sequence branch on natural-order data.
__global__ void k_conv_silu(const float* __restrict__ xz, const float* __restrict__ cw,
                            const float* __restrict__ cb, float* __restrict__ xhf,
                            _Float16* __restrict__ xhh, int nseq, int L, int dir)
{
    int i = blockIdx.x * blockDim.x + threadIdx.x;
    if (i >= nseq * L * 128) return;
    int d = i & 127, r = i >> 7;
    int b = r / L, l = r % L;
    float acc = cb[d];
    if (dir == 0) {
        for (int tap = 0; tap < 4; ++tap) {
            int ls = l - 3 + tap;
            if (ls >= 0) acc += xz[((size_t)(b * L + ls)) * 256 + d] * cw[d * 4 + tap];
        }
    } else {
        for (int tap = 0; tap < 4; ++tap) {
            int ls = l + tap;
            if (ls < L) acc += xz[((size_t)(b * L + ls)) * 256 + d] * cw[d * 4 + 3 - tap];
        }
    }
    float v = acc * sigmoidf_(acc);
    xhf[(size_t)r * 128 + d] = v;
    xhh[(size_t)r * 128 + d] = (_Float16)v;
}

// ---------------------------------------------------------------------------
// Selective scan. Lane = channel d; h[d,0:16] in VGPRs; shared (dt4,B16,C16)
// broadcast through LDS. delta = softplus(dt@dt_w+dt_b); D-skip; silu(z) gate.
// ---------------------------------------------------------------------------

// single-pass variant (pixel path: 25 independent sequences of 576)
__global__ void k_scan(const float* __restrict__ dbl, const float* __restrict__ xhf,
                       const float* __restrict__ xz, const float* __restrict__ dtw,
                       const float* __restrict__ dtb, const float* __restrict__ Alog,
                       const float* __restrict__ Dv, _Float16* __restrict__ yout,
                       int L, int dir)
{
    int b = blockIdx.x;
    int d = threadIdx.x;            // 0..127
    __shared__ float sh[36];
    float A[16];
#pragma unroll
    for (int n = 0; n < 16; ++n) A[n] = -__expf(Alog[d * 16 + n]);
    float w0 = dtw[d * 4 + 0], w1 = dtw[d * 4 + 1], w2 = dtw[d * 4 + 2], w3 = dtw[d * 4 + 3];
    float bb = dtb[d], Dd = Dv[d];
    float h[16];
#pragma unroll
    for (int n = 0; n < 16; ++n) h[n] = 0.f;
    for (int t = 0; t < L; ++t) {
        int l = dir ? (L - 1 - t) : t;
        size_t r = (size_t)b * L + l;
        if (d < 36) sh[d] = dbl[r * 48 + d];
        __syncthreads();
        float delta = softplusf_(sh[0] * w0 + sh[1] * w1 + sh[2] * w2 + sh[3] * w3 + bb);
        float u = xhf[r * 128 + d];
        float du = delta * u;
        float y = 0.f;
#pragma unroll
        for (int n = 0; n < 16; ++n) {
            h[n] = h[n] * __expf(delta * A[n]) + du * sh[4 + n];
            y += h[n] * sh[20 + n];
        }
        float z = xz[r * 256 + 128 + d];
        float out = (y + u * Dd) * (z * sigmoidf_(z));
        yout[r * 128 + d] = (_Float16)out;
        __syncthreads();
    }
}

// Chunked exact parallel scan for the window path (1 sequence of 14400).
// h_t = a_t h_{t-1} + b_t is linear in the initial state, so:
//  p1: per chunk, scan from h=0 -> chunk-end state hloc and decay product P
//  comb: S_c = hloc_c + P_c * S_{c-1}; hinit_c = S_{c-1}  (2048 lanes, 60 steps)
//  p2: re-scan each chunk from its exact hinit and emit gated outputs.
__global__ void k_scan_p1(const float* __restrict__ dbl, const float* __restrict__ xhf,
                          const float* __restrict__ dtw, const float* __restrict__ dtb,
                          const float* __restrict__ Alog,
                          float* __restrict__ hloc, float* __restrict__ Pp, int dir)
{
    int cchunk = blockIdx.x;
    int d = threadIdx.x;
    __shared__ float sh[36];
    float A[16];
#pragma unroll
    for (int n = 0; n < 16; ++n) A[n] = -__expf(Alog[d * 16 + n]);
    float w0 = dtw[d * 4 + 0], w1 = dtw[d * 4 + 1], w2 = dtw[d * 4 + 2], w3 = dtw[d * 4 + 3];
    float bb = dtb[d];
    float h[16], P[16];
#pragma unroll
    for (int n = 0; n < 16; ++n) { h[n] = 0.f; P[n] = 1.f; }
    for (int tt = 0; tt < LCH; ++tt) {
        int tau = cchunk * LCH + tt;
        size_t r = (size_t)(dir ? (RTOT - 1 - tau) : tau);
        if (d < 36) sh[d] = dbl[r * 48 + d];
        __syncthreads();
        float delta = softplusf_(sh[0] * w0 + sh[1] * w1 + sh[2] * w2 + sh[3] * w3 + bb);
        float du = delta * xhf[r * 128 + d];
#pragma unroll
        for (int n = 0; n < 16; ++n) {
            float a = __expf(delta * A[n]);
            h[n] = h[n] * a + du * sh[4 + n];
            P[n] *= a;
        }
        __syncthreads();
    }
#pragma unroll
    for (int n = 0; n < 16; ++n) {
        hloc[((size_t)cchunk * 128 + d) * 16 + n] = h[n];
        Pp[((size_t)cchunk * 128 + d) * 16 + n] = P[n];
    }
}

__global__ void k_scan_comb(const float* __restrict__ hloc, const float* __restrict__ Pp,
                            float* __restrict__ hinit)
{
    int i = blockIdx.x * blockDim.x + threadIdx.x;   // (d,n) pair, 2048 total
    if (i >= 2048) return;
    float s = 0.f;
    for (int c = 0; c < NCH; ++c) {
        hinit[(size_t)c * 2048 + i] = s;
        s = hloc[(size_t)c * 2048 + i] + Pp[(size_t)c * 2048 + i] * s;
    }
}

__global__ void k_scan_p2(const float* __restrict__ dbl, const float* __restrict__ xhf,
                          const float* __restrict__ xz, const float* __restrict__ dtw,
                          const float* __restrict__ dtb, const float* __restrict__ Alog,
                          const float* __restrict__ Dv, const float* __restrict__ hinit,
                          _Float16* __restrict__ yout, int dir)
{
    int cchunk = blockIdx.x;
    int d = threadIdx.x;
    __shared__ float sh[36];
    float A[16];
#pragma unroll
    for (int n = 0; n < 16; ++n) A[n] = -__expf(Alog[d * 16 + n]);
    float w0 = dtw[d * 4 + 0], w1 = dtw[d * 4 + 1], w2 = dtw[d * 4 + 2], w3 = dtw[d * 4 + 3];
    float bb = dtb[d], Dd = Dv[d];
    float h[16];
#pragma unroll
    for (int n = 0; n < 16; ++n) h[n] = hinit[(size_t)cchunk * 2048 + d * 16 + n];
    for (int tt = 0; tt < LCH; ++tt) {
        int tau = cchunk * LCH + tt;
        size_t r = (size_t)(dir ? (RTOT - 1 - tau) : tau);
        if (d < 36) sh[d] = dbl[r * 48 + d];
        __syncthreads();
        float delta = softplusf_(sh[0] * w0 + sh[1] * w1 + sh[2] * w2 + sh[3] * w3 + bb);
        float u = xhf[r * 128 + d];
        float du = delta * u;
        float y = 0.f;
#pragma unroll
        for (int n = 0; n < 16; ++n) {
            h[n] = h[n] * __expf(delta * A[n]) + du * sh[4 + n];
            y += h[n] * sh[20 + n];
        }
        float z = xz[r * 256 + 128 + d];
        float out = (y + u * Dd) * (z * sigmoidf_(z));
        yout[r * 128 + d] = (_Float16)out;
        __syncthreads();
    }
}

__global__ void k_add_h(const _Float16* __restrict__ a, const _Float16* __restrict__ b,
                        _Float16* __restrict__ o, int n)
{
    int i = blockIdx.x * blockDim.x + threadIdx.x;
    if (i < n) o[i] = (_Float16)((float)a[i] + (float)b[i]);
}

// pixel inverse shuffle + residual: t = lastx + scatter(yo)
__global__ void k_pixel_res(const float* __restrict__ lastx, const float* __restrict__ yo,
                            float* __restrict__ timg)
{
    int i = blockIdx.x * blockDim.x + threadIdx.x;
    if (i >= IMG_N) return;
    int w = i % HW, h = (i / HW) % HW, c = i / (HW * HW);
    int row = ((h % 5) * 5 + (w % 5)) * 576 + (h / 5) * 24 + (w / 5);
    timg[i] = lastx[i] + yo[(size_t)row * 64 + c];
}

// lastx += (window input t) + (window projection)  [= lastx + a]
__global__ void k_win_res(float* __restrict__ lastx, const float* __restrict__ timg,
                          const float* __restrict__ yo)
{
    int i = blockIdx.x * blockDim.x + threadIdx.x;
    if (i >= IMG_N) return;
    int l = i % RTOT, c = i / RTOT;
    lastx[i] = lastx[i] + timg[i] + yo[(size_t)l * 64 + c];
}

// hub-graph reduction: hub[c] = mean over the 450 hub rows (stride-32 nodes)
__global__ void k_hub(const float* __restrict__ h, float* __restrict__ hub, int C)
{
    int c = blockIdx.x * blockDim.x + threadIdx.x;
    if (c >= C) return;
    float s = 0.f;
    for (int k = 0; k < 450; ++k) s += h[(size_t)(k * 32) * C + c];
    hub[c] = s * (1.f / 450.f);
}

__global__ void k_gcn1(const float* __restrict__ h1, const float* __restrict__ hub,
                       const float* __restrict__ b1, _Float16* __restrict__ g1)
{
    int i = blockIdx.x * blockDim.x + threadIdx.x;
    if (i >= RTOT * 128) return;
    int c = i & 127, m = i >> 7;
    float v = (((m & 31) == 0) ? hub[c] : h1[i]) + b1[c];
    g1[i] = (_Float16)fmaxf(v, 0.f);
}

__global__ void k_final(const float* __restrict__ h2, const float* __restrict__ hub2,
                        const float* __restrict__ b2, const float* __restrict__ x,
                        const float* __restrict__ lx, const float* __restrict__ lxT,
                        float* __restrict__ out)
{
    int i = blockIdx.x * blockDim.x + threadIdx.x;
    if (i >= IMG_N) return;
    int w = i % HW, h = (i / HW) % HW, c = i / (HW * HW);
    int m = h * HW + w;
    float g = (((m & 31) == 0) ? hub2[c] : h2[(size_t)m * 64 + c]) + b2[c];
    out[i] = g + x[i] + lx[i] + lxT[(c * HW + w) * HW + h];
}

// ---------------------------------------------------------------------------
// Host-side orchestration
// ---------------------------------------------------------------------------
struct MambaP {
    const _Float16 *inw_h, *xprojw_h;
    const float *convw, *convb, *dtw, *dtb, *Alog, *Dv;
};

static void run_mamba(const _Float16* xn_h, const MambaP& p, int nseq, int L,
                      float* xz, float* xhf, _Float16* xhh, float* dbl,
                      float* hloc, float* Pp, float* hinit,
                      _Float16* y0, _Float16* y1, hipStream_t st)
{
    const int R = nseq * L;
    for (int dir = 0; dir < 2; ++dir) {
        const _Float16* inw = p.inw_h + (size_t)dir * 256 * 64;
        const _Float16* xpw = p.xprojw_h + (size_t)dir * 48 * 128;
        const float* dw = p.dtw + dir * 128 * 4;
        const float* db_ = p.dtb + dir * 128;
        const float* Al = p.Alog + dir * 128 * 16;
        const float* Dv = p.Dv + dir * 128;
        _Float16* yo_h = dir ? y1 : y0;
        gemm_wmma<4, 64><<<dim3(R / 16, 4), 32, 0, st>>>(xn_h, inw, xz, R, 256, 0);
        k_conv_silu<<<(R * 128 + 255) / 256, 256, 0, st>>>(xz, p.convw + dir * 128 * 4,
                                                           p.convb + dir * 128, xhf, xhh,
                                                           nseq, L, dir);
        gemm_wmma<3, 128><<<dim3(R / 16, 1), 32, 0, st>>>(xhh, xpw, dbl, R, 48, 0);
        if (nseq == 1) {   // window: chunked exact parallel scan (60-way)
            k_scan_p1<<<NCH, 128, 0, st>>>(dbl, xhf, dw, db_, Al, hloc, Pp, dir);
            k_scan_comb<<<8, 256, 0, st>>>(hloc, Pp, hinit);
            k_scan_p2<<<NCH, 128, 0, st>>>(dbl, xhf, xz, dw, db_, Al, Dv, hinit, yo_h, dir);
        } else {           // pixel: 25 independent sequences, single pass
            k_scan<<<nseq, 128, 0, st>>>(dbl, xhf, xz, dw, db_, Al, Dv, yo_h, L, dir);
        }
    }
}

extern "C" void kernel_launch(void* const* d_in, const int* in_sizes, int n_in,
                              void* d_out, int out_size, void* d_ws, size_t ws_size,
                              hipStream_t stream)
{
    (void)in_sizes; (void)n_in; (void)out_size;
    // --- inputs (setup_inputs dict order, all f32) ---
    const float* x_in       = (const float*)d_in[0];
    const float* pix_in_w   = (const float*)d_in[1];
    const float* pix_conv_w = (const float*)d_in[2];
    const float* pix_conv_b = (const float*)d_in[3];
    const float* pix_xproj  = (const float*)d_in[4];
    const float* pix_dt_w   = (const float*)d_in[5];
    const float* pix_dt_b   = (const float*)d_in[6];
    const float* pix_A_log  = (const float*)d_in[7];
    const float* pix_Dv     = (const float*)d_in[8];
    const float* pix_ln_g   = (const float*)d_in[9];
    const float* pix_ln_b   = (const float*)d_in[10];
    const float* win_in_w   = (const float*)d_in[11];
    const float* win_conv_w = (const float*)d_in[12];
    const float* win_conv_b = (const float*)d_in[13];
    const float* win_xproj  = (const float*)d_in[14];
    const float* win_dt_w   = (const float*)d_in[15];
    const float* win_dt_b   = (const float*)d_in[16];
    const float* win_A_log  = (const float*)d_in[17];
    const float* win_Dv     = (const float*)d_in[18];
    const float* win_ln_g   = (const float*)d_in[19];
    const float* win_ln_b   = (const float*)d_in[20];
    const float* pix_out_w  = (const float*)d_in[21];
    const float* win_out_w  = (const float*)d_in[22];
    const float* win_out_bw = (const float*)d_in[23];
    const float* gcn_w1     = (const float*)d_in[24];
    const float* gcn_b1     = (const float*)d_in[25];
    const float* gcn_w2     = (const float*)d_in[26];
    const float* gcn_b2     = (const float*)d_in[27];

    // --- workspace bump allocator (256B aligned) ---
    char* base = (char*)d_ws;
    size_t off = 0;
    auto alloc = [&](size_t bytes) -> void* {
        void* p = base + off;
        off = (off + bytes + 255) & ~(size_t)255;
        return p;
    };
    _Float16* pix_inw_h   = (_Float16*)alloc(4 * 2 * 256 * 64 * 2);
    _Float16* win_inw_h   = (_Float16*)alloc(4 * 2 * 256 * 64 * 2);
    _Float16* pix_xprj_h  = (_Float16*)alloc(4 * 2 * 48 * 128 * 2);
    _Float16* win_xprj_h  = (_Float16*)alloc(4 * 2 * 48 * 128 * 2);
    _Float16* pix_outw_h  = (_Float16*)alloc(4 * 64 * 128 * 2);
    _Float16* win_outw_h  = (_Float16*)alloc(4 * 64 * 128 * 2);
    _Float16* win_outbw_h = (_Float16*)alloc(4 * 64 * 128 * 2);
    _Float16* w1_h        = (_Float16*)alloc(128 * 256 * 2);
    _Float16* w2_h        = (_Float16*)alloc(64 * 128 * 2);
    float* imgT   = (float*)alloc((size_t)IMG_N * 4);
    float* lastx0 = (float*)alloc((size_t)IMG_N * 4);
    float* lastxT = (float*)alloc((size_t)IMG_N * 4);
    float* t_img  = (float*)alloc((size_t)IMG_N * 4);
    _Float16* xn_h   = (_Float16*)alloc((size_t)RTOT * 64 * 2);
    float* xz        = (float*)alloc((size_t)RTOT * 256 * 4);
    float* xhf       = (float*)alloc((size_t)RTOT * 128 * 4);
    _Float16* xhh    = (_Float16*)alloc((size_t)RTOT * 128 * 2);
    float* dbl       = (float*)alloc((size_t)RTOT * 48 * 4);
    _Float16* y0h    = (_Float16*)alloc((size_t)RTOT * 128 * 2);
    _Float16* y1h    = (_Float16*)alloc((size_t)RTOT * 128 * 2);
    _Float16* ysum_h = (_Float16*)alloc((size_t)RTOT * 128 * 2);
    float* yo        = (float*)alloc((size_t)RTOT * 64 * 4);
    float* nf        = (float*)alloc((size_t)RTOT * 256 * 4);
    _Float16* nf_h   = (_Float16*)alloc((size_t)RTOT * 256 * 2);
    float* h1        = (float*)alloc((size_t)RTOT * 128 * 4);
    _Float16* g1_h   = (_Float16*)alloc((size_t)RTOT * 128 * 2);
    float* h2        = (float*)alloc((size_t)RTOT * 64 * 4);
    float* hub1      = (float*)alloc(128 * 4);
    float* hub2      = (float*)alloc(64 * 4);
    float* hloc      = (float*)alloc((size_t)NCH * 2048 * 4);
    float* Pp        = (float*)alloc((size_t)NCH * 2048 * 4);
    float* hinit     = (float*)alloc((size_t)NCH * 2048 * 4);
    if (off > ws_size) return;   // workspace too small: bail deterministically

    const int TB = 256;
    // --- one-time (per launch) weight conversions to f16 ---
    k_f32_to_f16<<<(4 * 2 * 256 * 64 + TB - 1) / TB, TB, 0, stream>>>(pix_in_w, pix_inw_h, 4 * 2 * 256 * 64);
    k_f32_to_f16<<<(4 * 2 * 256 * 64 + TB - 1) / TB, TB, 0, stream>>>(win_in_w, win_inw_h, 4 * 2 * 256 * 64);
    k_xproj_pad<<<(8 * 48 * 128 + TB - 1) / TB, TB, 0, stream>>>(pix_xproj, pix_xprj_h);
    k_xproj_pad<<<(8 * 48 * 128 + TB - 1) / TB, TB, 0, stream>>>(win_xproj, win_xprj_h);
    k_f32_to_f16<<<(4 * 64 * 128 + TB - 1) / TB, TB, 0, stream>>>(pix_out_w, pix_outw_h, 4 * 64 * 128);
    k_f32_to_f16<<<(4 * 64 * 128 + TB - 1) / TB, TB, 0, stream>>>(win_out_w, win_outw_h, 4 * 64 * 128);
    k_f32_to_f16<<<(4 * 64 * 128 + TB - 1) / TB, TB, 0, stream>>>(win_out_bw, win_outbw_h, 4 * 64 * 128);
    k_f32_to_f16<<<(128 * 256 + TB - 1) / TB, TB, 0, stream>>>(gcn_w1, w1_h, 128 * 256);
    k_f32_to_f16<<<(64 * 128 + TB - 1) / TB, TB, 0, stream>>>(gcn_w2, w2_h, 64 * 128);
    k_transpose_img<<<(IMG_N + TB - 1) / TB, TB, 0, stream>>>(x_in, imgT);

    // --- two branches: normal image (params i=0,1) and transposed (i=2,3) ---
    for (int br = 0; br < 2; ++br) {
        float* lastx = br ? lastxT : lastx0;
        hipMemcpyAsync(lastx, br ? (const float*)imgT : x_in, (size_t)IMG_N * 4,
                       hipMemcpyDeviceToDevice, stream);
        for (int s = 0; s < 2; ++s) {
            const int i = br * 2 + s;
            // ---- pixel stage ----
            k_ln_pixel<<<(RTOT + TB - 1) / TB, TB, 0, stream>>>(lastx, pix_ln_g + i * 64,
                                                                pix_ln_b + i * 64, xn_h);
            MambaP mp{pix_inw_h + (size_t)i * 2 * 256 * 64, pix_xprj_h + (size_t)i * 2 * 48 * 128,
                      pix_conv_w + (size_t)i * 2 * 128 * 4, pix_conv_b + (size_t)i * 2 * 128,
                      pix_dt_w + (size_t)i * 2 * 128 * 4, pix_dt_b + (size_t)i * 2 * 128,
                      pix_A_log + (size_t)i * 2 * 128 * 16, pix_Dv + (size_t)i * 2 * 128};
            run_mamba(xn_h, mp, 25, 576, xz, xhf, xhh, dbl, hloc, Pp, hinit, y0h, y1h, stream);
            k_add_h<<<(RTOT * 128 + TB - 1) / TB, TB, 0, stream>>>(y0h, y1h, ysum_h, RTOT * 128);
            gemm_wmma<4, 128><<<dim3(RTOT / 16, 1), 32, 0, stream>>>(
                ysum_h, pix_outw_h + (size_t)i * 64 * 128, yo, RTOT, 64, 0);
            k_pixel_res<<<(IMG_N + TB - 1) / TB, TB, 0, stream>>>(lastx, yo, t_img);
            // ---- window stage ----
            k_ln_window<<<(RTOT + TB - 1) / TB, TB, 0, stream>>>(t_img, win_ln_g + i * 64,
                                                                 win_ln_b + i * 64, xn_h);
            MambaP mw{win_inw_h + (size_t)i * 2 * 256 * 64, win_xprj_h + (size_t)i * 2 * 48 * 128,
                      win_conv_w + (size_t)i * 2 * 128 * 4, win_conv_b + (size_t)i * 2 * 128,
                      win_dt_w + (size_t)i * 2 * 128 * 4, win_dt_b + (size_t)i * 2 * 128,
                      win_A_log + (size_t)i * 2 * 128 * 16, win_Dv + (size_t)i * 2 * 128};
            run_mamba(xn_h, mw, 1, RTOT, xz, xhf, xhh, dbl, hloc, Pp, hinit, y0h, y1h, stream);
            k_add_h<<<(RTOT * 128 + TB - 1) / TB, TB, 0, stream>>>(y0h, y1h, ysum_h, RTOT * 128);
            gemm_wmma<4, 128><<<dim3(RTOT / 16, 1), 32, 0, stream>>>(
                ysum_h, win_outw_h + (size_t)i * 64 * 128, yo, RTOT, 64, 0);
            if (s == 1) {  // capture node features xM / yM into nf (cols br*128 .. +127)
                gemm_wmma<4, 128><<<dim3(RTOT / 16, 1), 32, 0, stream>>>(
                    y0h, win_outw_h + (size_t)i * 64 * 128, nf + br * 128, RTOT, 256, 0);
                gemm_wmma<4, 128><<<dim3(RTOT / 16, 1), 32, 0, stream>>>(
                    y1h, win_outbw_h + (size_t)i * 64 * 128, nf + br * 128 + 64,
                    RTOT, 256, 1);  // reversed row store (yM)
            }
            k_win_res<<<(IMG_N + TB - 1) / TB, TB, 0, stream>>>(lastx, t_img, yo);
        }
    }

    // --- GCN (hub-collapsed) ---
    k_f32_to_f16<<<(RTOT * 256 + TB - 1) / TB, TB, 0, stream>>>(nf, nf_h, RTOT * 256);
    gemm_wmma<4, 256><<<dim3(RTOT / 16, 2), 32, 0, stream>>>(nf_h, w1_h, h1, RTOT, 128, 0);
    k_hub<<<1, 128, 0, stream>>>(h1, hub1, 128);
    k_gcn1<<<(RTOT * 128 + TB - 1) / TB, TB, 0, stream>>>(h1, hub1, gcn_b1, g1_h);
    gemm_wmma<4, 128><<<dim3(RTOT / 16, 1), 32, 0, stream>>>(g1_h, w2_h, h2, RTOT, 64, 0);
    k_hub<<<1, 64, 0, stream>>>(h2, hub2, 64);
    k_final<<<(IMG_N + TB - 1) / TB, TB, 0, stream>>>(h2, hub2, gcn_b2, x_in, lastx0, lastxT,
                                                      (float*)d_out);
}
